// MemoryEfficientMultiHeadCrossAttention_68023692034236
// MI455X (gfx1250) — matrix operand order repaired and checked by reading, not compiled, not run on hardware
//
#include <hip/hip_runtime.h>
#include <hip/hip_bf16.h>
#include <stdint.h>

// ---------------------------------------------------------------------------
// MI455X (gfx1250) bf16-WMMA cross-attention pipeline, async-LDS GEMM staging.
// D_MODEL=1024, N_HEADS=16, D_K=64, B=2, Sq=Skv=2048, tokens M=4096.
// ---------------------------------------------------------------------------

typedef __attribute__((ext_vector_type(16))) __bf16 v16bf;
typedef __attribute__((ext_vector_type(8)))  float  v8f;
typedef __attribute__((ext_vector_type(4)))  int    v4i;

union FragAB { v16bf v; uint32_t u[8]; uint4 u4[2]; };
union FragC  { v8f   v; float    f[8]; };

__device__ __forceinline__ uint32_t f2bf1(float x) {
  uint32_t u = __float_as_uint(x);
  return (u + 0x7fffu + ((u >> 16) & 1u)) >> 16;   // RNE fp32 -> bf16 bits
}
__device__ __forceinline__ uint32_t packbf(float a, float b) {
  return f2bf1(a) | (f2bf1(b) << 16);
}
__device__ __forceinline__ v8f wmma_bf16(v16bf a, v16bf b, v8f c) {
  // (neg_a, A, neg_b, B, c_mod, C, reuse_a, reuse_b)
  return __builtin_amdgcn_wmma_f32_16x16x32_bf16(false, a, false, b, (short)0, c, false, false);
}

// ---- async global->LDS copy (ASYNCcnt path), with sync fallback ----------
#if defined(__has_builtin)
#  if __has_builtin(__builtin_amdgcn_global_load_async_to_lds_b128)
#    define USE_ASYNC_LDS 1
#  endif
#endif
#ifndef USE_ASYNC_LDS
#  define USE_ASYNC_LDS 0
#endif

#if USE_ASYNC_LDS
typedef __attribute__((address_space(1))) v4i* as1_v4i;
typedef __attribute__((address_space(3))) v4i* as3_v4i;
#endif

__device__ __forceinline__ void copy16_g2l(const void* g, void* l) {
#if USE_ASYNC_LDS
  // flat-LDS low 32 bits == LDS offset (ISA 10.2)
  __builtin_amdgcn_global_load_async_to_lds_b128(
      (as1_v4i)(uintptr_t)g, (as3_v4i)(uintptr_t)l, 0, 0);
#else
  *reinterpret_cast<uint4*>(l) = *reinterpret_cast<const uint4*>(g);
#endif
}
__device__ __forceinline__ void async_wait3() {   // allow 3 in flight
#if USE_ASYNC_LDS
#  if __has_builtin(__builtin_amdgcn_s_wait_asynccnt)
  __builtin_amdgcn_s_wait_asynccnt(3);
#  else
  asm volatile("s_wait_asynccnt 0x3" ::: "memory");
#  endif
#endif
}
__device__ __forceinline__ void async_wait0() {
#if USE_ASYNC_LDS
#  if __has_builtin(__builtin_amdgcn_s_wait_asynccnt)
  __builtin_amdgcn_s_wait_asynccnt(0);
#  else
  asm volatile("s_wait_asynccnt 0x0" ::: "memory");
#  endif
#endif
}

// ---------------------------------------------------------------------------
// fp32 -> bf16 bulk convert (one-time; everything downstream reads bf16)
// ---------------------------------------------------------------------------
__global__ __launch_bounds__(256) void cvt_kernel(
    const float* __restrict__ src, unsigned short* __restrict__ dst, int n4)
{
  const int i = blockIdx.x * 256 + threadIdx.x;
  if (i < n4) {
    const float4 f = reinterpret_cast<const float4*>(src)[i];
    uint2 o;
    o.x = packbf(f.x, f.y);
    o.y = packbf(f.z, f.w);
    reinterpret_cast<uint2*>(dst)[i] = o;
  }
}

// ---------------------------------------------------------------------------
// GEMM: C[M,N] = A[M,1024] * W[N,1024]^T + bias[N], A/W bf16, K=1024.
// Double-buffered LDS tiles filled with async b128 copies.
// LDS row stride 80B: 64B of data in 4 16B chunks + 16B pad -> conflict-free
// uint4 fragment reads that land exactly on WMMA A/B chunk boundaries.
// MODE 0: bf16 -> Q[b,h,s,d]   MODE 1: bf16 -> K[b,h,s,d] | V^T[b,h,d,s]
// MODE 2: fp32 -> acc + bias + residual (out-projection)
// ---------------------------------------------------------------------------
#define ABUF 10240              // 128 rows * 80B
#define BBUF 5120               // 64 rows * 80B
#define STG  (ABUF + BBUF)

template<int MODE>
__global__ __launch_bounds__(256) void gemm_kernel(
    const unsigned short* __restrict__ A, const unsigned short* __restrict__ W,
    const float* __restrict__ bias, const float* __restrict__ residual,
    unsigned short* __restrict__ o0, unsigned short* __restrict__ o1,
    float* __restrict__ of)
{
  __shared__ __align__(16) uint8_t smem[2 * STG];

  const int tid  = threadIdx.x;
  const int wave = tid >> 5, lane = tid & 31, hl = lane >> 4, ln = lane & 15;
  const int wm   = wave >> 1, wn = wave & 1;      // 4x2 wave grid
  const int m0   = blockIdx.y * 128;
  const int n0   = blockIdx.x * 64;

  FragC acc[2][2];
  #pragma unroll
  for (int i = 0; i < 2; ++i)
    #pragma unroll
    for (int j = 0; j < 2; ++j)
      #pragma unroll
      for (int v = 0; v < 8; ++v) acc[i][j].f[v] = 0.f;

  // stage one 128x32(A) + 64x32(B) bf16 tile into LDS buffer `sm`
  auto stage_tile = [&](int k0, uint8_t* sm) {
    #pragma unroll
    for (int p = 0; p < 2; ++p) {                 // A: 512 chunks / 256 thr
      const int id = p * 256 + tid;
      const int r = id >> 2, c = id & 3;          // 4 chunks of 8 bf16 per row
      copy16_g2l(A + (size_t)(m0 + r) * 1024 + k0 + c * 8,
                 sm + r * 80 + c * 16);
    }
    {                                             // B: 256 chunks / 256 thr
      const int r = tid >> 2, c = tid & 3;
      copy16_g2l(W + (size_t)(n0 + r) * 1024 + k0 + c * 8,
                 sm + ABUF + r * 80 + c * 16);
    }
  };

  stage_tile(0, smem);                            // prologue: tile 0 in flight

  for (int kt = 0; kt < 32; ++kt) {
    uint8_t* cur = smem + (kt & 1) * STG;
    uint8_t* nxt = smem + ((kt + 1) & 1) * STG;

    __syncthreads();                              // `nxt` free for overwrite
    if (kt + 1 < 32) { stage_tile((kt + 1) * 32, nxt); async_wait3(); }
    else             { async_wait0(); }
    __syncthreads();                              // tile kt complete everywhere

    FragAB a[2], bfr[2];
    #pragma unroll
    for (int i = 0; i < 2; ++i) {                 // A 16x32: chunks hl, 2+hl
      const int r = wm * 32 + i * 16 + ln;
      a[i].u4[0] = *reinterpret_cast<const uint4*>(cur + r * 80 + hl * 16);
      a[i].u4[1] = *reinterpret_cast<const uint4*>(cur + r * 80 + (2 + hl) * 16);
    }
    #pragma unroll
    for (int j = 0; j < 2; ++j) {                 // B 32x16: chunks hl, 2+hl
      const int n = wn * 32 + j * 16 + ln;
      bfr[j].u4[0] = *reinterpret_cast<const uint4*>(cur + ABUF + n * 80 + hl * 16);
      bfr[j].u4[1] = *reinterpret_cast<const uint4*>(cur + ABUF + n * 80 + (2 + hl) * 16);
    }
    #pragma unroll
    for (int i = 0; i < 2; ++i)
      #pragma unroll
      for (int j = 0; j < 2; ++j)
        acc[i][j].v = wmma_bf16(a[i].v, bfr[j].v, acc[i][j].v);
  }

  // ---- epilogue (C layout: VGPR v -> row v + 8*hl, col = ln) ----
  #pragma unroll
  for (int i = 0; i < 2; ++i) {
    #pragma unroll
    for (int j = 0; j < 2; ++j) {
      #pragma unroll
      for (int v = 0; v < 8; ++v) {
        const int m = m0 + wm * 32 + i * 16 + v + 8 * hl;
        const int n = n0 + wn * 32 + j * 16 + ln;
        const float val = acc[i][j].f[v] + bias[n];
        if constexpr (MODE == 0) {
          const int b = m >> 11, s = m & 2047, h = n >> 6, d = n & 63;
          o0[(((size_t)(b * 16 + h)) * 2048 + s) * 64 + d] = (unsigned short)f2bf1(val);
        } else if constexpr (MODE == 1) {
          const int b = m >> 11, s = m & 2047;
          if (n < 1024) {
            const int h = n >> 6, d = n & 63;
            o0[(((size_t)(b * 16 + h)) * 2048 + s) * 64 + d] = (unsigned short)f2bf1(val);
          } else {
            const int n2 = n - 1024, h = n2 >> 6, d = n2 & 63;
            o1[(((size_t)(b * 16 + h)) * 64 + d) * 2048 + s] = (unsigned short)f2bf1(val);
          }
        } else {
          of[(size_t)m * 1024 + n] = val + residual[(size_t)m * 1024 + n];
        }
      }
    }
  }
}

// ---------------------------------------------------------------------------
// Flash attention: grid (Sq/128, B*H), 256 threads = 8 waves, 16 q-rows/wave.
// Q resident in registers; K [b,h,s,d] and V^T [b,h,d,s] feed B-fragments as
// contiguous 128-bit global loads; P transposed via per-wave padded LDS.
// ---------------------------------------------------------------------------
__global__ __launch_bounds__(256) void attn_kernel(
    const unsigned short* __restrict__ Qb, const unsigned short* __restrict__ Kb,
    const unsigned short* __restrict__ Vt, unsigned short* __restrict__ Ctx)
{
  __shared__ __align__(16) unsigned short Pld[8 * 16 * 72]; // stride 72 -> conflict-free
  const int tid  = threadIdx.x;
  const int wave = tid >> 5, lane = tid & 31, hl = lane >> 4, ln = lane & 15;
  const int bh   = blockIdx.y;
  const int b    = bh >> 4, h = bh & 15;
  const int q0   = blockIdx.x * 128 + wave * 16;
  unsigned short* P = &Pld[wave * 16 * 72];

  FragAB qa[2];
  {
    const unsigned short* qp = Qb + ((size_t)bh * 2048 + q0 + ln) * 64;
    #pragma unroll
    for (int f = 0; f < 2; ++f) {
      qa[f].u4[0] = *reinterpret_cast<const uint4*>(qp + f * 32 + hl * 8);
      qa[f].u4[1] = *reinterpret_cast<const uint4*>(qp + f * 32 + 16 + hl * 8);
    }
  }

  float rmax[8], rsum[8];
  FragC acc[4];
  #pragma unroll
  for (int v = 0; v < 8; ++v) { rmax[v] = -3.0e38f; rsum[v] = 0.f; }
  #pragma unroll
  for (int j = 0; j < 4; ++j)
    #pragma unroll
    for (int v = 0; v < 8; ++v) acc[j].f[v] = 0.f;

  const float scale = 0.125f;   // 1/sqrt(64)

  for (int kt = 0; kt < 32; ++kt) {
    const int key0 = kt * 64;

    FragC s[4];
    #pragma unroll
    for (int jt = 0; jt < 4; ++jt)
      #pragma unroll
      for (int v = 0; v < 8; ++v) s[jt].f[v] = 0.f;

    #pragma unroll
    for (int jt = 0; jt < 4; ++jt) {
      const unsigned short* kp = Kb + ((size_t)bh * 2048 + key0 + jt * 16 + ln) * 64;
      #pragma unroll
      for (int f = 0; f < 2; ++f) {
        FragAB kf;
        kf.u4[0] = *reinterpret_cast<const uint4*>(kp + f * 32 + hl * 8);
        kf.u4[1] = *reinterpret_cast<const uint4*>(kp + f * 32 + 16 + hl * 8);
        s[jt].v = wmma_bf16(qa[f].v, kf.v, s[jt].v);
      }
    }

    // online softmax: each row lives in one 16-lane half -> shfl_xor reduce
    #pragma unroll
    for (int v = 0; v < 8; ++v) {
      float t0 = s[0].f[v] * scale, t1 = s[1].f[v] * scale;
      float t2 = s[2].f[v] * scale, t3 = s[3].f[v] * scale;
      float tm = fmaxf(fmaxf(t0, t1), fmaxf(t2, t3));
      #pragma unroll
      for (int m = 1; m < 16; m <<= 1) tm = fmaxf(tm, __shfl_xor(tm, m, 32));
      const float nm   = fmaxf(rmax[v], tm);
      const float resc = __expf(rmax[v] - nm);
      rmax[v] = nm;
      const float p0 = __expf(t0 - nm), p1 = __expf(t1 - nm);
      const float p2 = __expf(t2 - nm), p3 = __expf(t3 - nm);
      float ts = p0 + p1 + p2 + p3;
      #pragma unroll
      for (int m = 1; m < 16; m <<= 1) ts += __shfl_xor(ts, m, 32);
      rsum[v] = rsum[v] * resc + ts;
      #pragma unroll
      for (int j = 0; j < 4; ++j) acc[j].f[v] *= resc;
      const int rv = v + 8 * hl;
      P[rv * 72 + 0 * 16 + ln] = (unsigned short)f2bf1(p0);
      P[rv * 72 + 1 * 16 + ln] = (unsigned short)f2bf1(p1);
      P[rv * 72 + 2 * 16 + ln] = (unsigned short)f2bf1(p2);
      P[rv * 72 + 3 * 16 + ln] = (unsigned short)f2bf1(p3);
    }
    // per-wave private LDS region; DS ops of one wave are in-order (no barrier)

    #pragma unroll
    for (int f = 0; f < 2; ++f) {
      FragAB pa;
      pa.u4[0] = *reinterpret_cast<const uint4*>(&P[ln * 72 + f * 32 + hl * 8]);
      pa.u4[1] = *reinterpret_cast<const uint4*>(&P[ln * 72 + f * 32 + 16 + hl * 8]);
      #pragma unroll
      for (int j = 0; j < 4; ++j) {
        const unsigned short* vp =
            Vt + ((size_t)bh * 64 + j * 16 + ln) * 2048 + key0 + f * 32 + hl * 8;
        FragAB vf;
        vf.u4[0] = *reinterpret_cast<const uint4*>(vp);
        vf.u4[1] = *reinterpret_cast<const uint4*>(vp + 16);
        acc[j].v = wmma_bf16(pa.v, vf.v, acc[j].v);
      }
    }
  }

  #pragma unroll
  for (int v = 0; v < 8; ++v) {
    const float inv = 1.0f / rsum[v];
    const int rv = v + 8 * hl;
    const size_t tok = (size_t)b * 2048 + q0 + rv;
    #pragma unroll
    for (int j = 0; j < 4; ++j)
      Ctx[tok * 1024 + h * 64 + j * 16 + ln] = (unsigned short)f2bf1(acc[j].f[v] * inv);
  }
}

// ---------------------------------------------------------------------------
// LayerNorm over rows of 1024 (residual already added by gemm MODE 2)
// ---------------------------------------------------------------------------
__global__ __launch_bounds__(256) void ln_kernel(
    const float* __restrict__ X, const float* __restrict__ gamma,
    const float* __restrict__ beta, float* __restrict__ out)
{
  const int row  = blockIdx.x;
  const int tid  = threadIdx.x;
  const int wave = tid >> 5, lane = tid & 31;
  __shared__ float ssum[8], ssq[8];

  float xv[4], sum = 0.f, sq = 0.f;
  #pragma unroll
  for (int k = 0; k < 4; ++k) {
    const float x = X[(size_t)row * 1024 + tid + k * 256];
    xv[k] = x; sum += x; sq += x * x;
  }
  #pragma unroll
  for (int m = 1; m < 32; m <<= 1) {
    sum += __shfl_xor(sum, m, 32);
    sq  += __shfl_xor(sq,  m, 32);
  }
  if (lane == 0) { ssum[wave] = sum; ssq[wave] = sq; }
  __syncthreads();
  sum = 0.f; sq = 0.f;
  #pragma unroll
  for (int w = 0; w < 8; ++w) { sum += ssum[w]; sq += ssq[w]; }
  const float mean = sum * (1.f / 1024.f);
  const float var  = sq * (1.f / 1024.f) - mean * mean;
  const float rstd = rsqrtf(var + 1e-5f);
  #pragma unroll
  for (int k = 0; k < 4; ++k) {
    const int c = tid + k * 256;
    out[(size_t)row * 1024 + c] = (xv[k] - mean) * rstd * gamma[c] + beta[c];
  }
}

// ---------------------------------------------------------------------------
extern "C" void kernel_launch(void* const* d_in, const int* in_sizes, int n_in,
                              void* d_out, int out_size, void* d_ws, size_t ws_size,
                              hipStream_t stream)
{
  (void)in_sizes; (void)n_in; (void)out_size; (void)ws_size;
  const float* query     = (const float*)d_in[0];
  const float* key_value = (const float*)d_in[1];
  const float* W_qkv     = (const float*)d_in[2];
  const float* b_qkv     = (const float*)d_in[3];
  const float* W_o       = (const float*)d_in[4];
  const float* b_o       = (const float*)d_in[5];
  const float* gamma     = (const float*)d_in[6];
  const float* beta      = (const float*)d_in[7];
  float* out = (float*)d_out;

  // ws: Qb 8 | Kb 8 | Vt 8 | Ctx 8 | Xq 8 | Xkv 8 | Wqkv 6 | Wo 2  = 56MB
  // AttnOut (16MB fp32) aliases Xq+Xkv, which are dead by the out-projection.
  char* ws = (char*)d_ws;
  const size_t MB = 1024 * 1024;
  unsigned short* Qb     = (unsigned short*)(ws);
  unsigned short* Kb     = (unsigned short*)(ws + 8 * MB);
  unsigned short* Vt     = (unsigned short*)(ws + 16 * MB);
  unsigned short* Ctx    = (unsigned short*)(ws + 24 * MB);
  unsigned short* Xq     = (unsigned short*)(ws + 32 * MB);
  unsigned short* Xkv    = (unsigned short*)(ws + 40 * MB);
  float*          AttnOut = (float*)        (ws + 32 * MB);   // aliases Xq/Xkv
  unsigned short* Wqkv_b = (unsigned short*)(ws + 48 * MB);
  unsigned short* Wo_b   = (unsigned short*)(ws + 54 * MB);

  const dim3 blk(256);
  // one-time fp32 -> bf16 conversions
  cvt_kernel<<<dim3(4096), blk, 0, stream>>>(query,     Xq,     4096 * 256);
  cvt_kernel<<<dim3(4096), blk, 0, stream>>>(key_value, Xkv,    4096 * 256);
  cvt_kernel<<<dim3(3072), blk, 0, stream>>>(W_qkv,     Wqkv_b, 3072 * 256);
  cvt_kernel<<<dim3(1024), blk, 0, stream>>>(W_o,       Wo_b,   1024 * 256);

  // Q = query @ Wq^T + bq          -> Q[b,h,s,d] bf16
  gemm_kernel<0><<<dim3(16, 32), blk, 0, stream>>>(
      Xq, Wqkv_b, b_qkv, nullptr, Qb, nullptr, nullptr);
  // K,V = key_value @ Wk/Wv^T + b  -> K[b,h,s,d], V^T[b,h,d,s] bf16
  gemm_kernel<1><<<dim3(32, 32), blk, 0, stream>>>(
      Xkv, Wqkv_b + 1024 * 1024, b_qkv + 1024, nullptr, Kb, Vt, nullptr);
  // flash attention -> Ctx[token,1024] bf16
  attn_kernel<<<dim3(16, 32), blk, 0, stream>>>(Qb, Kb, Vt, Ctx);
  // out = Ctx @ Wo^T + bo + query  -> fp32 (residual fused)
  gemm_kernel<2><<<dim3(16, 32), blk, 0, stream>>>(
      Ctx, Wo_b, b_o, query, nullptr, nullptr, AttnOut);
  // layernorm -> d_out
  ln_kernel<<<dim3(4096), blk, 0, stream>>>(AttnOut, gamma, beta, out);
}
